// MyModel_15788299780794
// MI455X (gfx1250) — matrix operand order repaired and checked by reading, not compiled, not run on hardware
//
#include <hip/hip_runtime.h>
#include <hip/hip_bf16.h>
#include <math.h>

typedef __attribute__((ext_vector_type(16))) _Float16 v16h;
typedef __attribute__((ext_vector_type(8)))  float    v8f;

#define BB    128
#define TT    512
#define BT    (BB*TT)        // 65536
#define EE    300
#define EP    320            // E padded to mult of 32
#define HHID  300
#define HP    320            // H padded
#define G4    1200           // 4*H
#define GBOTH 2400           // both directions
#define KK    9
#define MT    (BT/16)        // 4096 M-tiles
#define KT_E  (EP/32)        // 10
#define NT_IH (GBOTH/16)     // 150
#define KT_H  (HP/32)        // 10
#define NT_HH (G4/16)        // 75

#define WMMA16(a, b, c) __builtin_amdgcn_wmma_f32_16x16x32_f16(false, (a), false, (b), (short)0, (c), false, false)

// ---------------- weight packing into WMMA B-register image ----------------
// B matrix 32x16 (KxN), f16: lanes<16: slots0-3->K 0..7, slots4-7->K 16..23 ; lanes>=16: +8 on K groups ; N = lane&15
__global__ void pack_ih_kernel(const float* __restrict__ Wf, const float* __restrict__ Wb,
                               _Float16* __restrict__ Bpk) {
  int id = blockIdx.x * blockDim.x + threadIdx.x;
  const int total = KT_E * NT_IH * 512;
  if (id >= total) return;
  int e = id & 511, chunk = id >> 9;
  int nt = chunk % NT_IH, kt = chunk / NT_IH;
  int lane = e >> 4, pos = e & 15, slot = pos >> 1, hi = pos & 1;
  int n = nt * 16 + (lane & 15);
  int kbase = (lane < 16) ? 0 : 8;
  int k = kt * 32 + kbase + ((slot < 4) ? slot * 2 : 8 + slot * 2) + hi;          // embedding dim index
  const float* W = (n < G4) ? Wf : Wb;
  int g = (n < G4) ? n : n - G4;
  float v = (k < EE) ? W[g * EE + k] : 0.0f;        // B[k][n] = Wih[g][e]
  Bpk[id] = (_Float16)v;
}

__global__ void pack_hh_kernel(const float* __restrict__ Wf, const float* __restrict__ Wb,
                               _Float16* __restrict__ Bpk) {
  int id = blockIdx.x * blockDim.x + threadIdx.x;
  const int total = 2 * KT_H * NT_HH * 512;
  if (id >= total) return;
  int e = id & 511, chunk = id >> 9;
  const int per = KT_H * NT_HH;
  int dir = chunk / per, c2 = chunk % per;
  int nt = c2 % NT_HH, kt = c2 / NT_HH;
  int lane = e >> 4, pos = e & 15, slot = pos >> 1, hi = pos & 1;
  int n = nt * 16 + (lane & 15);                    // gate column g
  int kbase = (lane < 16) ? 0 : 8;
  int k = kt * 32 + kbase + ((slot < 4) ? slot * 2 : 8 + slot * 2) + hi;          // hidden index j
  const float* W = dir ? Wb : Wf;
  float v = (k < HHID) ? W[n * HHID + k] : 0.0f;    // B[j][g] = Whh[g][j]
  Bpk[id] = (_Float16)v;
}

__global__ void pack_bias_kernel(const float* __restrict__ bf1, const float* __restrict__ bf2,
                                 const float* __restrict__ bb1, const float* __restrict__ bb2,
                                 float* __restrict__ bias) {
  int n = blockIdx.x * blockDim.x + threadIdx.x;
  if (n >= GBOTH) return;
  bias[n] = (n < G4) ? (bf1[n] + bf2[n]) : (bb1[n - G4] + bb2[n - G4]);
}

// ---------------- embedding gather, packed into WMMA A-register image ------
// A 16x32 f16: lanes<16 (M=lane): slots0-3->K 0..7, slots4-7->K 16..23 ; lanes>=16: +8 on K groups
__global__ void embed_pack_kernel(const int* __restrict__ x, const float* __restrict__ emb,
                                  _Float16* __restrict__ Apk) {
  int id = blockIdx.x * blockDim.x + threadIdx.x;   // total 20,971,520 < 2^31
  int e = id & 511, chunk = id >> 9;
  int kt = chunk % KT_E, mt = chunk / KT_E;
  int lane = e >> 4, pos = e & 15, slot = pos >> 1, hi = pos & 1;
  int m = mt * 16 + (lane & 15);                    // flat (b*T + t) row
  int khalf = (lane < 16) ? 0 : 8;
  int base = (slot < 4) ? (slot * 2) : (16 + (slot - 4) * 2);
  int k = kt * 32 + base + khalf + hi;
  int tok = x[m];
  float v = (k < EE) ? emb[(size_t)tok * EE + k] : 0.0f;
  Apk[id] = (_Float16)v;
}

// ---------------- big GEMM: xg = emb @ [Wih_f;Wih_b]^T + bias --------------
// 1x2 register blocking per wave; dual accumulators break the WMMA RAW chain.
__global__ void gemm_xg_kernel(const _Float16* __restrict__ Apk, const _Float16* __restrict__ Bpk,
                               const float* __restrict__ bias, float* __restrict__ xg) {
  int lane = threadIdx.x & 31;
  int w = threadIdx.x >> 5;
  int nt0 = blockIdx.y * 16 + w * 2;                // even
  int mt = blockIdx.x;
  if (nt0 >= NT_IH) return;                         // wave-uniform
  int nt1 = nt0 + 1;                                // < NT_IH (150 even)
  v8f acc0a = {}, acc0b = {}, acc1a = {}, acc1b = {};
  const _Float16* ap = Apk + (size_t)mt * KT_E * 512 + lane * 16;
  const _Float16* bp = Bpk + lane * 16;
#pragma unroll
  for (int kt = 0; kt < KT_E; ++kt) {
    v16h a  = *reinterpret_cast<const v16h*>(ap + (size_t)kt * 512);
    v16h b0 = *reinterpret_cast<const v16h*>(bp + ((size_t)kt * NT_IH + nt0) * 512);
    v16h b1 = *reinterpret_cast<const v16h*>(bp + ((size_t)kt * NT_IH + nt1) * 512);
    if (kt & 1) { acc0b = WMMA16(a, b0, acc0b); acc1b = WMMA16(a, b1, acc1b); }
    else        { acc0a = WMMA16(a, b0, acc0a); acc1a = WMMA16(a, b1, acc1a); }
  }
  int n = lane & 15;
  int mbase = (lane < 16) ? 0 : 8;
  int g0 = nt0 * 16 + n, g1 = nt1 * 16 + n;
  float bs0 = bias[g0], bs1 = bias[g1];
#pragma unroll
  for (int v = 0; v < 8; ++v) {
    int m = mbase + v;
    size_t row = (size_t)(mt * 16 + m) * GBOTH;
    xg[row + g0] = acc0a[v] + acc0b[v] + bs0;
    xg[row + g1] = acc1a[v] + acc1b[v] + bs1;
  }
}

// ---------------- recurrent LSTM: one workgroup per (batch-group, dir) -----
// 320 threads = 10 waves. h kept in LDS in packed-A image; c in registers.
__global__ void lstm_kernel(const _Float16* __restrict__ BpkHH, const float* __restrict__ xg,
                            float* __restrict__ hf, float* __restrict__ hb) {
  __shared__ __align__(32) _Float16 ApkL[KT_H * 512]; // 10 KB, packed h (f16)
  __shared__ _Float16 gbufH[16 * G4];                  // 37.5 KB gate pre-activations
  int tid = threadIdx.x;                               // 0..319
  int lane = tid & 31, w = tid >> 5;                   // 10 waves
  int dir = blockIdx.y;
  int bg = blockIdx.x;                                 // 0..7 (16 batch rows each)
  const _Float16* Whh = BpkHH + (size_t)dir * (KT_H * NT_HH * 512);
  float* hout = dir ? hb : hf;

  for (int i = tid; i < KT_H * 512; i += 320) ApkL[i] = (_Float16)0.0f; // h0 = 0
  float creg[15];
#pragma unroll
  for (int r = 0; r < 15; ++r) creg[r] = 0.0f;         // c0 = 0
  __syncthreads();

  for (int step = 0; step < TT; ++step) {
    int t = dir ? (TT - 1 - step) : step;
    // prefetch next step's xg slice (16 rows x 4800B) toward the WGP
    if (step + 1 < TT) {
      int tn = dir ? (TT - 1 - (step + 1)) : (step + 1);
#pragma unroll
      for (int r = 0; r < 2; ++r) {
        int u = tid + 320 * r;                         // 0..639, need < 608
        int pm = u / 38, poff = (u % 38) * 128;        // 38*128 >= 4800
        if (pm < 16) {
          size_t prow = (size_t)(bg * 16 + pm) * TT + tn;
          __builtin_prefetch((const char*)(xg + prow * GBOTH + dir * G4) + poff, 0, 0);
        }
      }
    }
    // gates = h_prev @ Whh^T + xg[:, t, :]
    for (int nt = w; nt < NT_HH; nt += 10) {
      v8f acca = {}, accb = {};
#pragma unroll
      for (int kt = 0; kt < KT_H; kt += 2) {
        v16h a0 = *reinterpret_cast<const v16h*>(&ApkL[kt * 512 + lane * 16]);
        v16h b0 = *reinterpret_cast<const v16h*>(Whh + ((size_t)kt * NT_HH + nt) * 512 + lane * 16);
        acca = WMMA16(a0, b0, acca);
        v16h a1 = *reinterpret_cast<const v16h*>(&ApkL[(kt + 1) * 512 + lane * 16]);
        v16h b1 = *reinterpret_cast<const v16h*>(Whh + ((size_t)(kt + 1) * NT_HH + nt) * 512 + lane * 16);
        accb = WMMA16(a1, b1, accb);
      }
      int n = lane & 15;
      int mbase = (lane < 16) ? 0 : 8;
      int gcol = nt * 16 + n;
#pragma unroll
      for (int v = 0; v < 8; ++v) {
        int m = mbase + v;
        size_t row = (size_t)(bg * 16 + m) * TT + t;
        float g = acca[v] + accb[v] + xg[row * GBOTH + dir * G4 + gcol];
        gbufH[m * G4 + gcol] = (_Float16)g;
      }
    }
    __syncthreads();
    // pointwise cell; each thread owns 15 fixed (m,j) cells across all steps
#pragma unroll
    for (int r = 0; r < 15; ++r) {
      int u = tid + 320 * r;                           // < 4800
      int m = u / HHID, j = u % HHID;
      float gi = (float)gbufH[m * G4 + j];
      float gf = (float)gbufH[m * G4 + HHID + j];
      float gg = (float)gbufH[m * G4 + 2 * HHID + j];
      float go = (float)gbufH[m * G4 + 3 * HHID + j];
      float i_ = 1.0f / (1.0f + expf(-gi));
      float f_ = 1.0f / (1.0f + expf(-gf));
      float o_ = 1.0f / (1.0f + expf(-go));
      float c = f_ * creg[r] + i_ * tanhf(gg);
      creg[r] = c;
      float h = o_ * tanhf(c);
      size_t row = (size_t)(bg * 16 + m) * TT + t;
      hout[row * HHID + j] = h;
      // repack h into A-register image for next step
      int kt = j >> 5, kl = j & 31;
      int rr = kl & 15;
      int lane2 = (rr < 8) ? m : (m + 16);
      int kk = kl & 7;
      int slot = ((kl < 16) ? 0 : 4) + (kk >> 1);
      ApkL[kt * 512 + lane2 * 16 + slot * 2 + (kk & 1)] = (_Float16)h;
    }
    __syncthreads();
  }
}

// ---------------- logits -> softmax (K=9), one block per (b,t) -------------
__global__ void logits_softmax_kernel(const float* __restrict__ hf, const float* __restrict__ hb,
                                      const float* __restrict__ Wlin, const float* __restrict__ blin,
                                      float* __restrict__ out) {
  __shared__ float sm[KK][64];
  __shared__ float lg[KK];
  int bt = blockIdx.x;
  int tid = threadIdx.x; // 64
  float acc[KK];
#pragma unroll
  for (int k = 0; k < KK; ++k) acc[k] = 0.0f;
  const float* hfp = hf + (size_t)bt * HHID;
  const float* hbp = hb + (size_t)bt * HHID;
  for (int j = tid; j < HHID; j += 64) {
    float a = hfp[j], b = hbp[j];
#pragma unroll
    for (int k = 0; k < KK; ++k)
      acc[k] += a * Wlin[k * (2 * HHID) + j] + b * Wlin[k * (2 * HHID) + HHID + j];
  }
#pragma unroll
  for (int k = 0; k < KK; ++k) sm[k][tid] = acc[k];
  __syncthreads();
  if (tid < KK) {
    float s = blin[tid];
    for (int i = 0; i < 64; ++i) s += sm[tid][i];
    lg[tid] = s;
  }
  __syncthreads();
  if (tid == 0) {
    float mx = lg[0];
    for (int k = 1; k < KK; ++k) mx = fmaxf(mx, lg[k]);
    float ssum = 0.0f; float ex[KK];
    for (int k = 0; k < KK; ++k) { ex[k] = expf(lg[k] - mx); ssum += ex[k]; }
    float inv = 1.0f / ssum;
    for (int k = 0; k < KK; ++k) out[(size_t)bt * KK + k] = ex[k] * inv;
  }
}

// ---------------- CRF forward algorithm + numerator, one block per b -------
__global__ void crf_forward_kernel(const float* __restrict__ probs, const int* __restrict__ y,
                                   const float* __restrict__ start_t, const float* __restrict__ end_t,
                                   const float* __restrict__ trans, float* __restrict__ llh) {
  __shared__ float transS[81], startS[9], endS[9];
  __shared__ float alpha[9], em[9];
  int b = blockIdx.x, lane = threadIdx.x; // 32
  for (int i = lane; i < 81; i += 32) transS[i] = trans[i];
  if (lane < 9) { startS[lane] = start_t[lane]; endS[lane] = end_t[lane]; }
  __syncthreads();
  if (lane < 9) em[lane] = probs[((size_t)b * TT) * KK + lane];
  __syncthreads();
  if (lane < 9) alpha[lane] = startS[lane] + em[lane];
  float num = 0.0f; int prevTag = 0, lastTag = 0;
  if (lane == 0) {
    int t0 = y[b * TT];                                  // lengths >= 1 -> valid
    num = startS[t0] + em[t0];
    prevTag = t0; lastTag = t0;
  }
  __syncthreads();
  for (int t = 1; t < TT; ++t) {
    int yt = y[b * TT + t];
    bool msk = (yt != -1);
    if (lane < 9) em[lane] = probs[((size_t)b * TT + t) * KK + lane];
    __syncthreads();
    float nxt = 0.0f;
    if (lane < 9) {
      float mx = -1e30f;
#pragma unroll
      for (int i = 0; i < 9; ++i) mx = fmaxf(mx, alpha[i] + transS[i * 9 + lane]);
      float s = 0.0f;
#pragma unroll
      for (int i = 0; i < 9; ++i) s += expf(alpha[i] + transS[i * 9 + lane] - mx);
      nxt = mx + logf(s) + em[lane];
    }
    __syncthreads();
    if (msk && lane < 9) alpha[lane] = nxt;
    if (msk && lane == 0) {
      num += transS[prevTag * 9 + yt] + em[yt];
      prevTag = yt; lastTag = yt;
    }
    __syncthreads();
  }
  if (lane == 0) {
    num += endS[lastTag];
    float mx = -1e30f;
    for (int i = 0; i < 9; ++i) mx = fmaxf(mx, alpha[i] + endS[i]);
    float s = 0.0f;
    for (int i = 0; i < 9; ++i) s += expf(alpha[i] + endS[i] - mx);
    llh[b] = num - (mx + logf(s));
  }
}

__global__ void crf_reduce_kernel(const float* __restrict__ llh, float* __restrict__ out) {
  __shared__ float s[128];
  int tid = threadIdx.x;
  s[tid] = llh[tid];
  __syncthreads();
  for (int off = 64; off > 0; off >>= 1) {
    if (tid < off) s[tid] += s[tid + off];
    __syncthreads();
  }
  if (tid == 0) out[(size_t)BT * KK] = -s[0];
}

// ---------------- host launch ----------------------------------------------
extern "C" void kernel_launch(void* const* d_in, const int* in_sizes, int n_in,
                              void* d_out, int out_size, void* d_ws, size_t ws_size,
                              hipStream_t stream) {
  (void)in_sizes; (void)n_in; (void)out_size; (void)ws_size;
  const int*   x      = (const int*)d_in[0];
  const int*   y      = (const int*)d_in[1];
  const float* emb    = (const float*)d_in[2];
  const float* Wih_f  = (const float*)d_in[3];
  const float* Whh_f  = (const float*)d_in[4];
  const float* bih_f  = (const float*)d_in[5];
  const float* bhh_f  = (const float*)d_in[6];
  const float* Wih_b  = (const float*)d_in[7];
  const float* Whh_b  = (const float*)d_in[8];
  const float* bih_b  = (const float*)d_in[9];
  const float* bhh_b  = (const float*)d_in[10];
  const float* Wlin   = (const float*)d_in[11];
  const float* blin   = (const float*)d_in[12];
  const float* startt = (const float*)d_in[13];
  const float* endt   = (const float*)d_in[14];
  const float* trans  = (const float*)d_in[15];
  float* out = (float*)d_out;

  char* ws = (char*)d_ws;
  constexpr size_t SZ_APK  = (size_t)MT * KT_E * 512 * 2;            // 41,943,040
  constexpr size_t SZ_BIH  = (size_t)KT_E * NT_IH * 512 * 2;         //  1,536,000
  constexpr size_t SZ_BIAS = 9728;                                    // 2400*4 padded
  constexpr size_t SZ_BHH  = (size_t)2 * KT_H * NT_HH * 512 * 2;     //  1,536,000
  constexpr size_t SZ_XG   = (size_t)BT * GBOTH * 4;                 // 629 MB
  constexpr size_t SZ_H    = (size_t)BT * HHID * 4;                  // 78.6 MB each
  constexpr size_t OFF_APK  = 0;
  constexpr size_t OFF_BIH  = OFF_APK + SZ_APK;
  constexpr size_t OFF_BIAS = OFF_BIH + SZ_BIH;
  constexpr size_t OFF_BHH  = OFF_BIAS + SZ_BIAS;
  constexpr size_t OFF_XG   = OFF_BHH + SZ_BHH;
  constexpr size_t OFF_HF   = OFF_XG + SZ_XG;
  constexpr size_t OFF_HB   = OFF_HF + SZ_H;
  constexpr size_t OFF_CRF  = OFF_HB + SZ_H;

  _Float16* Apk    = (_Float16*)(ws + OFF_APK);
  _Float16* BpkIH  = (_Float16*)(ws + OFF_BIH);
  float*    biasIH = (float*)(ws + OFF_BIAS);
  _Float16* BpkHH  = (_Float16*)(ws + OFF_BHH);
  float*    xg     = (float*)(ws + OFF_XG);
  float*    hfp    = (float*)(ws + OFF_HF);
  float*    hbp    = (float*)(ws + OFF_HB);
  float*    crf    = (float*)(ws + OFF_CRF);

  pack_ih_kernel<<<(KT_E * NT_IH * 512 + 255) / 256, 256, 0, stream>>>(Wih_f, Wih_b, BpkIH);
  pack_bias_kernel<<<(GBOTH + 255) / 256, 256, 0, stream>>>(bih_f, bhh_f, bih_b, bhh_b, biasIH);
  pack_hh_kernel<<<(2 * KT_H * NT_HH * 512 + 255) / 256, 256, 0, stream>>>(Whh_f, Whh_b, BpkHH);
  embed_pack_kernel<<<(MT * KT_E * 512) / 256, 256, 0, stream>>>(x, emb, Apk);

  dim3 g1(MT, (NT_IH + 15) / 16);
  gemm_xg_kernel<<<g1, 256, 0, stream>>>(Apk, BpkIH, biasIH, xg);

  dim3 g2(BB / 16, 2);
  lstm_kernel<<<g2, 320, 0, stream>>>(BpkHH, xg, hfp, hbp);

  logits_softmax_kernel<<<BT, 64, 0, stream>>>(hfp, hbp, Wlin, blin, out);
  crf_forward_kernel<<<BB, 32, 0, stream>>>(out, y, startt, endt, trans, crf);
  crf_reduce_kernel<<<1, 128, 0, stream>>>(crf, out);
}
